// HRFA_9225589752078
// MI455X (gfx1250) — matrix operand order repaired and checked by reading, not compiled, not run on hardware
//
#include <hip/hip_runtime.h>
#include <hip/hip_bf16.h>
#include <math.h>

// ---------------------------------------------------------------------------
// Problem constants (from reference): B=8, DIM=128, H=W=128.
// ---------------------------------------------------------------------------
#define BATCH 8
#define HW    16384      // 128*128
#define WID   128
#define HEI   128

typedef __attribute__((ext_vector_type(16))) __bf16 v16bf;
typedef __attribute__((ext_vector_type(8)))  float  v8f;

// Native bf16 convert (gfx1250 has hardware f32->bf16, RNE).
__device__ __forceinline__ __bf16 f2bf(float f) { return (__bf16)f; }

// ---------------------------------------------------------------------------
// Channels-first LayerNorm, 4 pixels (float4) per thread, coalesced b128.
// ---------------------------------------------------------------------------
__global__ __launch_bounds__(256)
void ln_cf(const float* __restrict__ in, const float* __restrict__ g,
           const float* __restrict__ bt, float* __restrict__ out, int C) {
  int idx = blockIdx.x * 256 + threadIdx.x;            // over BATCH*HW/4
  int b = idx / (HW / 4), q = idx % (HW / 4);
  const float4* base = (const float4*)(in + (size_t)b * C * HW) + q;
  float sx = 0.f, sy = 0.f, sz = 0.f, sw = 0.f;
  float qx = 0.f, qy = 0.f, qz = 0.f, qw = 0.f;
  for (int c = 0; c < C; ++c) {
    float4 v = base[(size_t)c * (HW / 4)];
    sx += v.x; sy += v.y; sz += v.z; sw += v.w;
    qx += v.x * v.x; qy += v.y * v.y; qz += v.z * v.z; qw += v.w * v.w;
  }
  float inv = 1.0f / (float)C;
  float mx = sx * inv, my = sy * inv, mz = sz * inv, mw = sw * inv;
  float rx = rsqrtf(qx * inv - mx * mx + 1e-6f);
  float ry = rsqrtf(qy * inv - my * my + 1e-6f);
  float rz = rsqrtf(qz * inv - mz * mz + 1e-6f);
  float rw = rsqrtf(qw * inv - mw * mw + 1e-6f);
  float4* ob = (float4*)(out + (size_t)b * C * HW) + q;
  for (int c = 0; c < C; ++c) {
    float4 v = base[(size_t)c * (HW / 4)];
    float gc = g[c], bc = bt[c];
    float4 o;
    o.x = gc * ((v.x - mx) * rx) + bc;
    o.y = gc * ((v.y - my) * ry) + bc;
    o.z = gc * ((v.z - mz) * rz) + bc;
    o.w = gc * ((v.w - mw) * rw) + bc;
    ob[(size_t)c * (HW / 4)] = o;
  }
}

// ---------------------------------------------------------------------------
// Global average pool of the first 16 channels (float4 reads).
// ---------------------------------------------------------------------------
__global__ __launch_bounds__(256)
void pool16(const float* __restrict__ in, int C, float* __restrict__ outp) {
  int b = blockIdx.x >> 4, c = blockIdx.x & 15;
  const float4* base = (const float4*)(in + (size_t)b * C * HW + (size_t)c * HW);
  float s = 0.f;
  for (int p = threadIdx.x; p < HW / 4; p += 256) {
    float4 v = base[p];
    s += (v.x + v.y) + (v.z + v.w);
  }
  __shared__ float sh[256];
  sh[threadIdx.x] = s; __syncthreads();
  for (int st = 128; st > 0; st >>= 1) {
    if (threadIdx.x < st) sh[threadIdx.x] += sh[threadIdx.x + st];
    __syncthreads();
  }
  if (threadIdx.x == 0) outp[blockIdx.x] = sh[0] / (float)HW;
}

// ---------------------------------------------------------------------------
// Dynamic-kernel generator MLP (tiny): k = W2*gelu(W1*pool+b1)+b2.
// ---------------------------------------------------------------------------
__global__ __launch_bounds__(256)
void kgen(const float* __restrict__ pool,
          const float* __restrict__ w1, const float* __restrict__ b1,
          const float* __restrict__ w2, const float* __restrict__ b2,
          float* __restrict__ dynK) {
  __shared__ float h[BATCH * 8];
  int t = threadIdx.x;
  if (t < BATCH * 8) {
    int b = t / 8, j = t % 8;
    float s = b1[j];
    for (int c = 0; c < 16; ++c) s += w1[j * 16 + c] * pool[b * 16 + c];
    h[t] = 0.5f * s * (1.0f + erff(s * 0.70710678118654752f));  // exact gelu
  }
  __syncthreads();
  for (int i = t; i < BATCH * 144; i += 256) {
    int b = i / 144, o = i % 144;
    float s = b2[o];
    for (int j = 0; j < 8; ++j) s += w2[o * 8 + j] * h[b * 8 + j];
    dynK[i] = s;
  }
}

// ---------------------------------------------------------------------------
// Strided channel-slice copy (F_idt part of the concat), float4.
// ---------------------------------------------------------------------------
__global__ __launch_bounds__(256)
void copych(const float* __restrict__ in, int inC, int inOff,
            float* __restrict__ out, int outC, int outOff, int nc) {
  size_t idx = (size_t)blockIdx.x * 256 + threadIdx.x;   // over B*nc*HW/4
  size_t per = (size_t)nc * (HW / 4);
  int b = (int)(idx / per);
  size_t rem = idx % per;
  int c = (int)(rem / (HW / 4)), q = (int)(rem % (HW / 4));
  const float4* s = (const float4*)(in + (size_t)b * inC * HW + (size_t)(inOff + c) * HW) + q;
  float4* d = (float4*)(out + (size_t)b * outC * HW + (size_t)(outOff + c) * HW) + q;
  *d = *s;
}

// ---------------------------------------------------------------------------
// 1x1 conv as bf16 WMMA GEMM: out[b,o,p] = sum_k W[o,k]*in'[b,k,p] + bias[o].
// HASMUL (compile-time) fuses in' = in * mul -> no branches in the hot loop.
// Each wave owns a 32(M) x 32(N) macro-tile (2 M x 2 N 16-wide tiles); every
// A/B fragment feeds 2 WMMAs. Lane-half base pointers hoisted so all loads
// and stores use literal instruction offsets.
// ---------------------------------------------------------------------------
template <bool HASMUL>
__global__ __launch_bounds__(256)
void gemm1x1(const float* __restrict__ in, int inC, int inOff,
             const float* __restrict__ mul, int mulC, int mulOff,
             const float* __restrict__ w, const float* __restrict__ bias,
             float* __restrict__ out, int outC, int outOff,
             int Cin, int Cout) {
  int gw = (blockIdx.x * 256 + threadIdx.x) >> 5;
  int lane = threadIdx.x & 31;
  int mGrp = Cout >> 5, pTiles = HW >> 5;                // 32-wide M and N
  int total = BATCH * pTiles * mGrp;
  if (gw >= total) return;                               // exact grid anyway
  int mg = gw % mGrp;
  int pt = (gw / mGrp) % pTiles;
  int b  = gw / (mGrp * pTiles);
  bool lo = lane < 16;
  int n = lane & 15;
  int p0 = pt << 5, moff = mg << 5;
  int kbA = lo ? 0 : 16;                                 // A lane-half K base
  int kbB = lo ? 0 : 16;                                 // B lane-half K base

  const float* inB  = in + (size_t)b * inC * HW + (size_t)inOff * HW + p0 + n;
  const float* mulB = HASMUL
      ? mul + (size_t)b * mulC * HW + (size_t)mulOff * HW + p0 + n : nullptr;
  const float* wRow0 = w + (size_t)(moff + n) * Cin + kbA;
  const float* wRow1 = w + (size_t)(moff + 16 + n) * Cin + kbA;

  v8f acc00 = {0.f,0.f,0.f,0.f,0.f,0.f,0.f,0.f};
  v8f acc01 = acc00, acc10 = acc00, acc11 = acc00;

  for (int k0 = 0; k0 < Cin; k0 += 32) {
    v16bf A0, A1;
    const float* pW0 = wRow0 + k0;
    const float* pW1 = wRow1 + k0;
#pragma unroll
    for (int e = 0; e < 16; ++e) {                       // A: 16x32 bf16 frags
      int off = e;                                       // literal offsets
      A0[e] = f2bf(pW0[off]);
      A1[e] = f2bf(pW1[off]);
    }
    const float* pk  = inB + (size_t)(k0 + kbB) * HW;    // literal offsets below
    const float* pmk = HASMUL ? mulB + (size_t)(k0 + kbB) * HW : nullptr;
#pragma unroll
    for (int t = 0; t < 2; ++t) {
      v16bf Bv;
#pragma unroll
      for (int e = 0; e < 16; ++e) {                     // B: 32x16 bf16 frag
        float v = pk[e * HW + t * 16];
        if (HASMUL) v *= pmk[e * HW + t * 16];
        Bv[e] = f2bf(v);
      }
      if (t == 0) {
        acc00 = __builtin_amdgcn_wmma_f32_16x16x32_bf16(false, A0, false, Bv,
                                                        (short)0, acc00, false, false);
        acc10 = __builtin_amdgcn_wmma_f32_16x16x32_bf16(false, A1, false, Bv,
                                                        (short)0, acc10, false, false);
      } else {
        acc01 = __builtin_amdgcn_wmma_f32_16x16x32_bf16(false, A0, false, Bv,
                                                        (short)0, acc01, false, false);
        acc11 = __builtin_amdgcn_wmma_f32_16x16x32_bf16(false, A1, false, Bv,
                                                        (short)0, acc11, false, false);
      }
    }
  }
  int oh = lo ? 0 : 8;
  float* outB = out + (size_t)b * outC * HW + (size_t)outOff * HW + p0 + n;
#pragma unroll
  for (int mi = 0; mi < 2; ++mi) {
    float* ob = outB + (size_t)(moff + mi * 16 + oh) * HW;
    const float* bb = bias + moff + mi * 16 + oh;
#pragma unroll
    for (int r = 0; r < 8; ++r) {                        // D: M=r+oh, N=n
      float brv = bb[r];
      if (mi == 0) {
        ob[r * HW]      = acc00[r] + brv;
        ob[r * HW + 16] = acc01[r] + brv;
      } else {
        ob[r * HW]      = acc10[r] + brv;
        ob[r * HW + 16] = acc11[r] + brv;
      }
    }
  }
}

// ---------------------------------------------------------------------------
// Dense 16->16 large-kernel conv (LK in {7,9,11}, compile-time) as implicit
// WMMA GEMM, fused with the per-sample dynamic depthwise 3x3 add.
// One wave per (batch,row,32-pixel strip); kx fully unrolled so weight and
// activation loads use literal offsets. Borders: unconditional loads into a
// guard-banded workspace + value select-0 (no address clamping, no exec
// divergence). K = 16 channels x 2 taps per WMMA (tap = lane half).
// ---------------------------------------------------------------------------
template <int LK>
__global__ __launch_bounds__(256)
void lkconv(const float* __restrict__ in, int inC,
            const float* __restrict__ lkw,
            const float* __restrict__ dynK,
            float* __restrict__ out, int outC) {
  constexpr int PAD = LK / 2;
  int gw = (blockIdx.x * 256 + threadIdx.x) >> 5;
  int lane = threadIdx.x & 31;
  if (gw >= BATCH * HEI * (WID / 32)) return;            // exact grid anyway
  int xt = gw & 3;
  int y  = (gw >> 2) & (HEI - 1);
  int b  = gw >> 9;
  int x0 = xt << 5;                                      // 32-pixel strip
  bool lo = lane < 16;
  int n = lane & 15;
  const float* inb = in + (size_t)b * inC * HW;          // F_att = ch 0..15
  // Per-lane weight base: row o=n, channel base 0.
  const float* wLane = lkw + ((size_t)n * 16) * (LK * LK);
  int kxoff = lo ? 0 : 1;                                // lane half selects tap

  v8f acc0 = {0.f,0.f,0.f,0.f,0.f,0.f,0.f,0.f};
  v8f acc1 = acc0;
  for (int ky = 0; ky < LK; ++ky) {
    int yy = y + ky - PAD;
    bool yok = (unsigned)yy < (unsigned)HEI;
    int yyc = yok ? yy : 0;                              // wave-uniform row base
    const float* pA = wLane + ky * LK;                   // + c8*LK*LK + kx + tap
    const float* colbase = inb + (size_t)yyc * WID + x0 + n - PAD + kxoff;
#pragma unroll
    for (int kx = 0; kx < LK; kx += 2) {
      constexpr int DUMMY = 0; (void)DUMMY;
      bool t1ok = (kx + 1) < LK;                         // compile-time
      v16bf A;
#pragma unroll
      for (int e = 0; e < 16; ++e) {
        // K layout: lane half -> tap (lo?0:1); c = e.
        int c8 = e, tap = lo ? 0 : 1;
        float wv = (tap == 0 || t1ok)
            ? pA[c8 * (LK * LK) + kx + tap] : 0.f;       // literal offsets
        A[e] = f2bf(wv);
      }
      bool tapok = lo || t1ok;
#pragma unroll
      for (int t = 0; t < 2; ++t) {
        int xx = x0 + t * 16 + n + kx + kxoff - PAD;
        bool ok = yok && (unsigned)xx < (unsigned)WID && tapok;
        const float* colp = colbase + t * 16 + kx;
        v16bf Bv;
#pragma unroll
        for (int e = 0; e < 16; ++e) {                   // c = e (literal offsets)
          float v = colp[e * HW];                        // unconditional load
          Bv[e] = f2bf(ok ? v : 0.f);
        }
        if (t == 0)
          acc0 = __builtin_amdgcn_wmma_f32_16x16x32_bf16(false, A, false, Bv,
                                                         (short)0, acc0, false, false);
        else
          acc1 = __builtin_amdgcn_wmma_f32_16x16x32_bf16(false, A, false, Bv,
                                                         (short)0, acc1, false, false);
      }
    }
  }
  // Fused dynamic depthwise 3x3 (per-sample kernels) + store.
  float* outb = out + (size_t)b * outC * HW;
  const float* dkb = dynK + b * 144;
#pragma unroll
  for (int t = 0; t < 2; ++t) {
    int x = x0 + t * 16 + n;
#pragma unroll
    for (int r = 0; r < 8; ++r) {
      int o = r + (lo ? 0 : 8);
      float s = t == 0 ? acc0[r] : acc1[r];
      const float* ib = inb + (size_t)o * HW + (size_t)y * WID + x;  // center
      const float* dk = dkb + o * 9;
#pragma unroll
      for (int dy = 0; dy < 3; ++dy) {
        bool yok = (unsigned)(y + dy - 1) < (unsigned)HEI;
#pragma unroll
        for (int dx = 0; dx < 3; ++dx) {
          bool xok = (unsigned)(x + dx - 1) < (unsigned)WID;
          float v = ib[(dy - 1) * WID + (dx - 1)];       // literal offsets
          s += (yok && xok) ? dk[dy * 3 + dx] * v : 0.f;
        }
      }
      outb[(size_t)o * HW + y * WID + x] = s;
    }
  }
}

// ---------------------------------------------------------------------------
// Depthwise 3x3 (32 ch) + bias + residual add, strided concat write.
// Borders via guard-banded loads + value select (literal offsets).
// ---------------------------------------------------------------------------
__global__ __launch_bounds__(256)
void dw3add(const float* __restrict__ in,                // standalone 32-ch tensor
            const float* __restrict__ w, const float* __restrict__ bias,
            const float* __restrict__ add, int addC, int addOff,
            float* __restrict__ out, int outC, int outOff) {
  size_t idx = (size_t)blockIdx.x * 256 + threadIdx.x;   // over B*32*HW
  int b = (int)(idx / (32 * HW));
  int c = (int)((idx / HW) % 32);
  int p = (int)(idx % HW);
  int y = p >> 7, x = p & 127;
  const float* base = in + ((size_t)b * 32 + c) * HW + p;  // center
  const float* wc = w + c * 9;
  float s = bias[c];
#pragma unroll
  for (int dy = 0; dy < 3; ++dy) {
    bool yok = (unsigned)(y + dy - 1) < (unsigned)HEI;
#pragma unroll
    for (int dx = 0; dx < 3; ++dx) {
      bool xok = (unsigned)(x + dx - 1) < (unsigned)WID;
      float v = base[(dy - 1) * WID + (dx - 1)];         // literal offsets
      s += (yok && xok) ? wc[dy * 3 + dx] * v : 0.f;
    }
  }
  s += add[(size_t)b * addC * HW + (size_t)(addOff + c) * HW + p];
  out[(size_t)b * outC * HW + (size_t)(outOff + c) * HW + p] = s;
}

// ---------------------------------------------------------------------------
// Orchestration.
// Input flattening assumption (setup_inputs() insertion order, depth-first):
//  0:x  1,2:norm1  3-9:a1{kg1_w,kg1_b,kg2_w,kg2_b,lk_w,fusion_w,fusion_b}
//  10,11:v1 12,13:v11 14,15:v12 16,17:conv3_1 18,19:norm2 20-26:a2
//  27,28:v2 29,30:v21 31,32:v22 33,34:proj2 35,36:conv3_2 37,38:norm3
//  39-45:a3 46,47:v3 48,49:v31 50,51:v32 52,53:proj3 54,55:conv3_3
// ---------------------------------------------------------------------------
extern "C" void kernel_launch(void* const* d_in, const int* in_sizes, int n_in,
                              void* d_out, int out_size, void* d_ws, size_t ws_size,
                              hipStream_t stream) {
  (void)in_sizes; (void)n_in; (void)out_size; (void)ws_size;
  const float** I = (const float**)d_in;
  float* out = (float*)d_out;

  // Guard band: border loads may over/under-run a row by <= WID+1 elements;
  // all such loads land inside [ws, ws+used) thanks to 1KB margins.
  const size_t GUARD = 256;                              // floats
  float* ws = (float*)d_ws + GUARD;

  // Workspace bump layout (floats), ~302 MB total.
  const size_t SZ128 = (size_t)BATCH * 128 * HW;
  const size_t SZ96  = (size_t)BATCH * 96  * HW;
  const size_t SZ32  = (size_t)BATCH * 32  * HW;
  float* XN   = ws;                   // LN(x), 128 ch (xs slices live here)
  float* Abuf = XN   + SZ128;         // attention output a (<=96 ch)
  float* Tbuf = Abuf + SZ96;          // concat(lk+dyn, F_idt); reused as v(x) tmp
  float* Xbuf = Tbuf + SZ96;          // X1 then X2 (pre-LN concat)
  float* XNs  = Xbuf + SZ96;          // X1N then X2N
  float* TMP2 = XNs  + SZ96;          // v12/v22/v32 output (32 ch)
  float* TMP3 = TMP2 + SZ32;          // proj output (32 ch)
  float* pool = TMP3 + SZ32;          // (B,16)
  float* dynK = pool + BATCH * 16;    // (B,144)

  const int LN_G = (BATCH * HW / 4) / 256;                       // 128
  const int LK_G = (BATCH * HEI * (WID / 32) * 32) / 256;        // 512
  const int DW_G = (BATCH * 32 * HW) / 256;                      // 16384
  auto gemmG = [](int Cout) { return (BATCH * (HW / 32) * (Cout / 32)) / 8; };
  auto cpyG  = [](int nc)   { return (BATCH * nc * (HW / 4)) / 256; };

  // ---- Stage 0: LN1 over 128 channels --------------------------------------
  ln_cf<<<LN_G, 256, 0, stream>>>(I[0], I[1], I[2], XN, 128);

  // ---- Stage 1: a1 on xs[0]=XN[0:32], lk=7 ---------------------------------
  pool16<<<BATCH * 16, 256, 0, stream>>>(XN, 128, pool);
  kgen<<<1, 256, 0, stream>>>(pool, I[3], I[4], I[5], I[6], dynK);
  lkconv<7><<<LK_G, 256, 0, stream>>>(XN, 128, I[7], dynK, Tbuf, 32);
  copych<<<cpyG(16), 256, 0, stream>>>(XN, 128, 16, Tbuf, 32, 16, 16);
  gemm1x1<false><<<gemmG(32), 256, 0, stream>>>(Tbuf, 32, 0, nullptr, 0, 0,
      I[8], I[9], Abuf, 32, 0, 32, 32);                                   // fusion
  gemm1x1<false><<<gemmG(32), 256, 0, stream>>>(XN, 128, 0, nullptr, 0, 0,
      I[10], I[11], Tbuf, 32, 0, 32, 32);                                 // v1
  gemm1x1<true><<<gemmG(32), 256, 0, stream>>>(Tbuf, 32, 0, Abuf, 32, 0,
      I[12], I[13], Xbuf, 64, 32, 32, 32);                                // v11 -> X1[32:64]
  gemm1x1<false><<<gemmG(32), 256, 0, stream>>>(XN, 128, 32, nullptr, 0, 0,
      I[14], I[15], TMP2, 32, 0, 32, 32);                                 // v12
  dw3add<<<DW_G, 256, 0, stream>>>(TMP2, I[16], I[17], Abuf, 32, 0,
                                   Xbuf, 64, 0);                          // +a -> X1[0:32]
  ln_cf<<<LN_G, 256, 0, stream>>>(Xbuf, I[18], I[19], XNs, 64);           // X1N

  // ---- Stage 2: a2 on X1N (64 ch), lk=9 ------------------------------------
  pool16<<<BATCH * 16, 256, 0, stream>>>(XNs, 64, pool);
  kgen<<<1, 256, 0, stream>>>(pool, I[20], I[21], I[22], I[23], dynK);
  lkconv<9><<<LK_G, 256, 0, stream>>>(XNs, 64, I[24], dynK, Tbuf, 64);
  copych<<<cpyG(48), 256, 0, stream>>>(XNs, 64, 16, Tbuf, 64, 16, 48);
  gemm1x1<false><<<gemmG(64), 256, 0, stream>>>(Tbuf, 64, 0, nullptr, 0, 0,
      I[25], I[26], Abuf, 64, 0, 64, 64);                                 // fusion
  gemm1x1<false><<<gemmG(64), 256, 0, stream>>>(XNs, 64, 0, nullptr, 0, 0,
      I[27], I[28], Tbuf, 64, 0, 64, 64);                                 // v2
  gemm1x1<true><<<gemmG(64), 256, 0, stream>>>(Tbuf, 64, 0, Abuf, 64, 0,
      I[29], I[30], Xbuf, 96, 32, 64, 64);                                // v21 -> X2[32:96]
  gemm1x1<false><<<gemmG(32), 256, 0, stream>>>(XN, 128, 64, nullptr, 0, 0,
      I[31], I[32], TMP2, 32, 0, 32, 32);                                 // v22
  gemm1x1<false><<<gemmG(32), 256, 0, stream>>>(Abuf, 64, 0, nullptr, 0, 0,
      I[33], I[34], TMP3, 32, 0, 64, 32);                                 // proj2
  dw3add<<<DW_G, 256, 0, stream>>>(TMP2, I[35], I[36], TMP3, 32, 0,
                                   Xbuf, 96, 0);                          // X2[0:32]
  ln_cf<<<LN_G, 256, 0, stream>>>(Xbuf, I[37], I[38], XNs, 96);           // X2N

  // ---- Stage 3: a3 on X2N (96 ch), lk=11 -----------------------------------
  pool16<<<BATCH * 16, 256, 0, stream>>>(XNs, 96, pool);
  kgen<<<1, 256, 0, stream>>>(pool, I[39], I[40], I[41], I[42], dynK);
  lkconv<11><<<LK_G, 256, 0, stream>>>(XNs, 96, I[43], dynK, Tbuf, 96);
  copych<<<cpyG(80), 256, 0, stream>>>(XNs, 96, 16, Tbuf, 96, 16, 80);
  gemm1x1<false><<<gemmG(96), 256, 0, stream>>>(Tbuf, 96, 0, nullptr, 0, 0,
      I[44], I[45], Abuf, 96, 0, 96, 96);                                 // fusion
  gemm1x1<false><<<gemmG(96), 256, 0, stream>>>(XNs, 96, 0, nullptr, 0, 0,
      I[46], I[47], Tbuf, 96, 0, 96, 96);                                 // v3
  gemm1x1<true><<<gemmG(96), 256, 0, stream>>>(Tbuf, 96, 0, Abuf, 96, 0,
      I[48], I[49], out, 128, 32, 96, 96);                                // v31 -> out[32:128]
  gemm1x1<false><<<gemmG(32), 256, 0, stream>>>(XN, 128, 96, nullptr, 0, 0,
      I[50], I[51], TMP2, 32, 0, 32, 32);                                 // v32
  gemm1x1<false><<<gemmG(32), 256, 0, stream>>>(Abuf, 96, 0, nullptr, 0, 0,
      I[52], I[53], TMP3, 32, 0, 96, 32);                                 // proj3
  dw3add<<<DW_G, 256, 0, stream>>>(TMP2, I[54], I[55], TMP3, 32, 0,
                                   out, 128, 0);                          // out[0:32]
}
